// MultiScaleAttention_20117626814753
// MI455X (gfx1250) — matrix operand order repaired and checked by reading, not compiled, not run on hardware
//
#include <hip/hip_runtime.h>

typedef __attribute__((ext_vector_type(16))) __bf16 v16bf;
typedef __attribute__((ext_vector_type(8)))  float  v8f;

#define BATCH   8
#define NTOK    3137          // 1 + 16*14*14
#define DIMC    768
#define NHEAD   8
#define HD      96
#define NQ      785           // 1 + 16*7*7
#define NQP     800           // padded to 50*16
#define NQTILES 50
#define NKV     393           // 1 + 8*7*7
#define KPAD    416           // padded to 13*32
#define M_QKV   (BATCH*NTOK)  // 25096
#define M_PROJ  (BATCH*NQ)    // 6280

// ---- gfx1250 async global->LDS path (probe via __has_builtin) -------------
#if defined(__has_builtin)
#  if __has_builtin(__builtin_amdgcn_global_load_async_to_lds_b128) && \
      __has_builtin(__builtin_amdgcn_s_wait_asynccnt)
#    define HAVE_ASYNC_LDS 1
#  endif
#endif
#ifndef HAVE_ASYNC_LDS
#  define HAVE_ASYNC_LDS 0
#endif

#if HAVE_ASYNC_LDS
typedef int b128vec __attribute__((vector_size(16)));
typedef __attribute__((address_space(1))) b128vec* g_b128_ptr;
typedef __attribute__((address_space(3))) b128vec* l_b128_ptr;
#endif

__device__ inline void copy16_g2l(void* lds, const void* g) {
#if HAVE_ASYNC_LDS
    __builtin_amdgcn_global_load_async_to_lds_b128(
        (g_b128_ptr)g, (l_b128_ptr)lds, 0, 0);
#else
    *(uint4*)lds = *(const uint4*)g;
#endif
}
__device__ inline void async_fence() {
#if HAVE_ASYNC_LDS
    __builtin_amdgcn_s_wait_asynccnt(0);
#endif
}

union Frag { v16bf v; uint4 q[2]; };
union Pack16 { __bf16 h[16]; uint4 q[2]; };

__device__ inline v8f wmma_bf16(const Frag& a, const Frag& b, v8f c) {
    return __builtin_amdgcn_wmma_f32_16x16x32_bf16(
        false, a.v, false, b.v, (short)0, c, false, false);
}

// ---------------------------------------------------------------------------
// fp32 -> bf16 conversion
// ---------------------------------------------------------------------------
__global__ void f2bf_kernel(const float* __restrict__ in, __bf16* __restrict__ out, int n) {
    int i = blockIdx.x * blockDim.x + threadIdx.x;
    if (i < n) out[i] = (__bf16)in[i];
}

// ---------------------------------------------------------------------------
// WMMA GEMM: C[M,N] = A[M,K] * Bw[N,K]^T + bias
// MODE 0: A fp32 (x), scatter-store bf16 into qkv layout [3][B][h][tok][96]
// MODE 1: A bf16 (attn out), store fp32 row-major into d_out
// Block 256 thr = 8 waves (4x2), tile 128x128x32, double-buffered LDS with
// async global->LDS staging when available.
// OOB A rows are clamped (not zeroed): row r of A only affects output row r,
// which is guarded at the store.
// ---------------------------------------------------------------------------
template <int MODE>
__global__ __launch_bounds__(256) void gemm_wmma_kernel(
    const void* __restrict__ Aptr, const __bf16* __restrict__ Bw,
    const float* __restrict__ bias, void* __restrict__ Optr,
    int M, int K)
{
    __shared__ __bf16 As[2][128 * 40];   // 32 + 8 pad
    __shared__ __bf16 Bs[2][128 * 40];

    const int tid  = threadIdx.x;
    const int lane = tid & 31;
    const int wid  = tid >> 5;
    const int wm   = wid >> 1;        // 0..3 -> 32 rows each
    const int wn   = wid & 1;        // 0..1 -> 64 cols each
    const int lrow = lane & 15;
    const int hi   = lane >> 4;
    const int m0   = blockIdx.y * 128;
    const int n0   = blockIdx.x * 128;
    const int ldr  = tid >> 1;        // 0..127
    const int ldc  = (tid & 1) * 16;  // 0 / 16
    const int KT   = K >> 5;

    int arow = m0 + ldr;
    if (arow >= M) arow = M - 1;      // clamp; garbage rows guarded at store

    const v8f vzero = {0.f,0.f,0.f,0.f,0.f,0.f,0.f,0.f};
    v8f acc[2][4];
#pragma unroll
    for (int s = 0; s < 2; s++)
#pragma unroll
        for (int t = 0; t < 4; t++) acc[s][t] = vzero;

    Pack16 areg;
    auto loadA_regs = [&](int k0) {
        const float* A = (const float*)Aptr;
        const float4* src = (const float4*)(A + (size_t)arow * K + k0 + ldc);
#pragma unroll
        for (int i = 0; i < 4; i++) {
            float4 f = src[i];
            areg.h[i*4+0] = (__bf16)f.x; areg.h[i*4+1] = (__bf16)f.y;
            areg.h[i*4+2] = (__bf16)f.z; areg.h[i*4+3] = (__bf16)f.w;
        }
    };
    auto storeA_regs = [&](int buf) {
        *(uint4*)&As[buf][ldr * 40 + ldc]     = areg.q[0];
        *(uint4*)&As[buf][ldr * 40 + ldc + 8] = areg.q[1];
    };
    auto stageA_async = [&](int k0, int buf) {
        const __bf16* g = (const __bf16*)Aptr + (size_t)arow * K + k0 + ldc;
        copy16_g2l(&As[buf][ldr * 40 + ldc],     g);
        copy16_g2l(&As[buf][ldr * 40 + ldc + 8], g + 8);
    };
    auto stageB = [&](int k0, int buf) {
        const __bf16* g = Bw + (size_t)(n0 + ldr) * K + k0 + ldc;
        copy16_g2l(&Bs[buf][ldr * 40 + ldc],     g);
        copy16_g2l(&Bs[buf][ldr * 40 + ldc + 8], g + 8);
    };

    // ---- prologue: stage k-tile 0 into buffer 0 ----
    if (MODE == 0) { loadA_regs(0); storeA_regs(0); }
    else          { stageA_async(0, 0); }
    stageB(0, 0);

    for (int kt = 0; kt < KT; ++kt) {
        const int cur = kt & 1, nxt = cur ^ 1;
        async_fence();
        __syncthreads();

        // issue next stage before computing the current one
        if (kt + 1 < KT) {
            if (MODE == 0) loadA_regs((kt + 1) * 32);
            else           stageA_async((kt + 1) * 32, nxt);
            stageB((kt + 1) * 32, nxt);
        }

        Frag a[2], b[4];
#pragma unroll
        for (int s = 0; s < 2; s++) {
            const __bf16* p = &As[cur][(wm * 32 + s * 16 + lrow) * 40];
            a[s].q[0] = *(const uint4*)(p + hi * 8);        // K 0..7 / 8..15
            a[s].q[1] = *(const uint4*)(p + 16 + hi * 8);   // K 16..23 / 24..31
        }
#pragma unroll
        for (int t = 0; t < 4; t++) {
            const __bf16* p = &Bs[cur][(wn * 64 + t * 16 + lrow) * 40];
            b[t].q[0] = *(const uint4*)(p + hi * 8);        // K 0..7 / 8..15
            b[t].q[1] = *(const uint4*)(p + 16 + hi * 8);   // K 16..23 / 24..31
        }
#pragma unroll
        for (int s = 0; s < 2; s++)
#pragma unroll
            for (int t = 0; t < 4; t++)
                acc[s][t] = wmma_bf16(a[s], b[t], acc[s][t]);

        if (MODE == 0 && kt + 1 < KT) storeA_regs(nxt);
    }

    // ---- epilogue ----
#pragma unroll
    for (int s = 0; s < 2; s++)
#pragma unroll
    for (int t = 0; t < 4; t++) {
        const int nn = n0 + wn * 64 + t * 16 + lrow;
        const float bv = bias[nn];
#pragma unroll
        for (int r = 0; r < 8; r++) {
            const int mm = m0 + wm * 32 + s * 16 + r + hi * 8;
            if (mm < M) {
                const float val = acc[s][t][r] + bv;
                if (MODE == 0) {
                    __bf16* O = (__bf16*)Optr;
                    const int which = nn / DIMC;
                    const int c     = nn - which * DIMC;
                    const int h     = c / HD;
                    const int d     = c - h * HD;
                    const int bidx  = mm / NTOK;
                    const int tok   = mm - bidx * NTOK;
                    O[((((size_t)which * BATCH + bidx) * NHEAD + h) * NTOK + tok) * HD + d]
                        = (__bf16)val;
                } else {
                    ((float*)Optr)[(size_t)mm * DIMC + nn] = val;
                }
            }
        }
    }
}

// ---------------------------------------------------------------------------
// Depthwise 3x3x3 conv pool (+cls passthrough) + LayerNorm over head dim.
// One 128-thread block per (bh, out-token). transposed=1 -> write [d][tok].
// ---------------------------------------------------------------------------
__global__ __launch_bounds__(128) void pool_ln_kernel(
    const __bf16* __restrict__ src,      // [64][NTOK][96]
    const float* __restrict__ cw,        // [96][27]
    const float* __restrict__ g, const float* __restrict__ beta,
    float* __restrict__ outF,            // nullable, [64][ntok_out][96]
    __bf16* __restrict__ outB,
    int stride_t, int ntok_out, int ldOut, int transposed)
{
    const int bh  = blockIdx.x / ntok_out;
    const int tok = blockIdx.x - bh * ntok_out;
    const int tid = threadIdx.x;
    const int d   = tid;

    const __bf16* s = src + (size_t)bh * NTOK * HD;
    float val = 0.f;
    if (d < HD) {
        if (tok == 0) {
            val = (float)s[d];
        } else {
            const int sp = tok - 1;
            const int x  = sp % 7;
            const int y  = (sp / 7) % 7;
            const int t  = sp / 49;
            const int t0 = t * stride_t - 1, y0 = y * 2 - 1, x0 = x * 2 - 1;
#pragma unroll
            for (int dt = 0; dt < 3; dt++)
#pragma unroll
            for (int dy = 0; dy < 3; dy++)
#pragma unroll
            for (int dx = 0; dx < 3; dx++) {
                const int ti = t0 + dt, yi = y0 + dy, xi = x0 + dx;
                if (ti >= 0 && ti < 16 && yi >= 0 && yi < 14 && xi >= 0 && xi < 14) {
                    const int lin = (ti * 14 + yi) * 14 + xi;
                    val += cw[d * 27 + (dt * 3 + dy) * 3 + dx]
                         * (float)s[(size_t)(1 + lin) * HD + d];
                }
            }
        }
    }
    __shared__ float rs[128], rq[128];
    rs[tid] = (d < HD) ? val : 0.f;
    rq[tid] = (d < HD) ? val * val : 0.f;
    __syncthreads();
    for (int sft = 64; sft > 0; sft >>= 1) {
        if (tid < sft) { rs[tid] += rs[tid + sft]; rq[tid] += rq[tid + sft]; }
        __syncthreads();
    }
    const float mu  = rs[0] * (1.f / HD);
    const float var = rq[0] * (1.f / HD) - mu * mu;
    if (d < HD) {
        const float yv = (val - mu) * rsqrtf(var + 1e-5f) * g[d] + beta[d];
        if (outF) outF[((size_t)bh * ntok_out + tok) * HD + d] = yv;
        if (transposed)
            outB[((size_t)bh * HD + d) * ldOut + tok] = (__bf16)yv;
        else
            outB[((size_t)bh * ldOut + tok) * HD + d] = (__bf16)yv;
    }
}

// ---------------------------------------------------------------------------
// Attention: one wave per (b,h,16-query tile). S=q*k^T (WMMA), fp32 softmax
// in LDS, O=P*v (WMMA via v^T), +q residual, store bf16 merged-head layout.
// ---------------------------------------------------------------------------
__global__ __launch_bounds__(32) void attn_kernel(
    const __bf16* __restrict__ qb,   // [64][NQP][96]
    const __bf16* __restrict__ kb,   // [64][KPAD][96]
    const __bf16* __restrict__ vt,   // [64][96][KPAD]
    const float*  __restrict__ qf,   // [64][NQ][96]
    __bf16* __restrict__ xout)       // [B][NQ][768]
{
    __shared__ float  Sld[16 * 400];
    __shared__ __bf16 Pld[16 * 416];

    const int bh   = blockIdx.x / NQTILES;
    const int qt   = blockIdx.x - bh * NQTILES;
    const int bidx = bh / NHEAD;
    const int h    = bh - bidx * NHEAD;
    const int lane = threadIdx.x;
    const int lrow = lane & 15;
    const int hi   = lane >> 4;
    const float scale = 0.10206207261596575f;   // 1/sqrt(96)

    const __bf16* qb_bh = qb + (size_t)bh * NQP * HD;
    const __bf16* kb_bh = kb + (size_t)bh * KPAD * HD;
    const __bf16* vt_bh = vt + (size_t)bh * HD * KPAD;
    const float*  qf_bh = qf + (size_t)bh * NQ * HD;

    // q A-fragments: 3 chunks over HD=96
    Frag aq[3];
    {
        const __bf16* qrow = qb_bh + (size_t)(qt * 16 + lrow) * HD;
#pragma unroll
        for (int c = 0; c < 3; c++) {
            aq[c].q[0] = *(const uint4*)(qrow + c * 32 + hi * 8);
            aq[c].q[1] = *(const uint4*)(qrow + c * 32 + 16 + hi * 8);
        }
    }

    const v8f vzero = {0.f,0.f,0.f,0.f,0.f,0.f,0.f,0.f};

    // ---- S = q * k^T, 25 key tiles ----
    for (int kt = 0; kt < 25; kt++) {
        v8f s = vzero;
        const __bf16* krow = kb_bh + (size_t)(kt * 16 + lrow) * HD;
#pragma unroll
        for (int c = 0; c < 3; c++) {
            Frag bk;
            bk.q[0] = *(const uint4*)(krow + c * 32 + hi * 8);
            bk.q[1] = *(const uint4*)(krow + c * 32 + 16 + hi * 8);
            s = wmma_bf16(aq[c], bk, s);
        }
#pragma unroll
        for (int r = 0; r < 8; r++)
            Sld[(r + hi * 8) * 400 + kt * 16 + lrow] = s[r] * scale;
    }
    __syncthreads();

    // ---- softmax: lane handles (row = lane&15, cols strided by 2) ----
    {
        const int row = lrow;
        float mx = -1e30f;
        for (int c = hi; c < NKV; c += 2) mx = fmaxf(mx, Sld[row * 400 + c]);
        mx = fmaxf(mx, __shfl_xor(mx, 16, 32));
        float sum = 0.f;
        for (int c = hi; c < NKV; c += 2) {
            const float p = __expf(Sld[row * 400 + c] - mx);
            Sld[row * 400 + c] = p;
            sum += p;
        }
        sum += __shfl_xor(sum, 16, 32);
        const float inv = 1.f / sum;
        for (int c = hi; c < KPAD; c += 2) {
            const float p = (c < NKV) ? Sld[row * 400 + c] * inv : 0.f;
            Pld[row * 416 + c] = (__bf16)p;
        }
    }
    __syncthreads();

    // ---- O = P * v  (13 K-chunks x 6 d-tiles) ----
    v8f o[6];
#pragma unroll
    for (int dt = 0; dt < 6; dt++) o[dt] = vzero;
    for (int kc = 0; kc < 13; kc++) {
        Frag ap;
        const __bf16* prow = &Pld[lrow * 416 + kc * 32];
        ap.q[0] = *(const uint4*)(prow + hi * 8);
        ap.q[1] = *(const uint4*)(prow + 16 + hi * 8);
#pragma unroll
        for (int dt = 0; dt < 6; dt++) {
            Frag bv;
            const __bf16* vrow = vt_bh + (size_t)(dt * 16 + lrow) * KPAD + kc * 32;
            bv.q[0] = *(const uint4*)(vrow + hi * 8);
            bv.q[1] = *(const uint4*)(vrow + 16 + hi * 8);
            o[dt] = wmma_bf16(ap, bv, o[dt]);
        }
    }

    // ---- +q residual, store merged-head bf16 ----
#pragma unroll
    for (int dt = 0; dt < 6; dt++)
#pragma unroll
    for (int r = 0; r < 8; r++) {
        const int m = qt * 16 + r + hi * 8;
        const int d = dt * 16 + lrow;
        if (m < NQ) {
            const float val = o[dt][r] + qf_bh[(size_t)m * HD + d];
            xout[((size_t)bidx * NQ + m) * DIMC + h * HD + d] = (__bf16)val;
        }
    }
}

// ---------------------------------------------------------------------------
extern "C" void kernel_launch(void* const* d_in, const int* in_sizes, int n_in,
                              void* d_out, int out_size, void* d_ws, size_t ws_size,
                              hipStream_t stream) {
    const float* x      = (const float*)d_in[0];
    const float* qkv_w  = (const float*)d_in[1];
    const float* qkv_b  = (const float*)d_in[2];
    const float* proj_w = (const float*)d_in[3];
    const float* proj_b = (const float*)d_in[4];
    const float* cqw    = (const float*)d_in[5];
    const float* ckw    = (const float*)d_in[6];
    const float* cvw    = (const float*)d_in[7];
    const float* lnq_g  = (const float*)d_in[8];
    const float* lnq_b  = (const float*)d_in[9];
    const float* lnk_g  = (const float*)d_in[10];
    const float* lnk_b  = (const float*)d_in[11];
    const float* lnv_g  = (const float*)d_in[12];
    const float* lnv_b  = (const float*)d_in[13];
    float* out = (float*)d_out;

    // workspace layout (bytes)
    constexpr size_t SLICE    = (size_t)BATCH * NHEAD * NTOK * HD;   // bf16 elems per q/k/v
    constexpr size_t SZ_QKV   = 3 * SLICE * 2;
    constexpr size_t SZ_QKVW  = (size_t)3 * DIMC * DIMC * 2;
    constexpr size_t SZ_PROJW = (size_t)DIMC * DIMC * 2;
    constexpr size_t SZ_QF    = (size_t)64 * NQ * HD * 4;
    constexpr size_t SZ_QB    = (size_t)64 * NQP * HD * 2;
    constexpr size_t SZ_KB    = (size_t)64 * KPAD * HD * 2;
    constexpr size_t SZ_VT    = (size_t)64 * HD * KPAD * 2;

    char* ws = (char*)d_ws;
    size_t off = 0;
    __bf16* qkvB   = (__bf16*)(ws + off); off += SZ_QKV;
    __bf16* qkvwB  = (__bf16*)(ws + off); off += SZ_QKVW;
    __bf16* projwB = (__bf16*)(ws + off); off += SZ_PROJW;
    float*  qpoolF = (float* )(ws + off); off += SZ_QF;
    __bf16* qB     = (__bf16*)(ws + off); off += SZ_QB;
    __bf16* kB     = (__bf16*)(ws + off); off += SZ_KB;
    __bf16* vtB    = (__bf16*)(ws + off); off += SZ_VT;
    __bf16* attnxB = (__bf16*)(ws + off); off += (size_t)BATCH * NQ * DIMC * 2;

    // 1) weight conversion
    {
        int n1 = 3 * DIMC * DIMC;
        f2bf_kernel<<<(n1 + 255) / 256, 256, 0, stream>>>(qkv_w, qkvwB, n1);
        int n2 = DIMC * DIMC;
        f2bf_kernel<<<(n2 + 255) / 256, 256, 0, stream>>>(proj_w, projwB, n2);
    }

    // 2) QKV GEMM (WMMA), scatter into per-head layout
    gemm_wmma_kernel<0><<<dim3((3 * DIMC) / 128, (M_QKV + 127) / 128), 256, 0, stream>>>(
        (const void*)x, qkvwB, qkv_b, (void*)qkvB, M_QKV, DIMC);

    // 3) zero padded attention operand buffers
    (void)hipMemsetAsync(qB, 0, SZ_QB, stream);
    (void)hipMemsetAsync(kB, 0, SZ_KB, stream);
    (void)hipMemsetAsync(vtB, 0, SZ_VT, stream);

    // 4) pool + LN for q, k, v
    pool_ln_kernel<<<64 * NQ, 128, 0, stream>>>(
        qkvB + 0 * SLICE, cqw, lnq_g, lnq_b, qpoolF, qB, 1, NQ, NQP, 0);
    pool_ln_kernel<<<64 * NKV, 128, 0, stream>>>(
        qkvB + 1 * SLICE, ckw, lnk_g, lnk_b, nullptr, kB, 2, NKV, KPAD, 0);
    pool_ln_kernel<<<64 * NKV, 128, 0, stream>>>(
        qkvB + 2 * SLICE, cvw, lnv_g, lnv_b, nullptr, vtB, 2, NKV, KPAD, 1);

    // 5) attention (WMMA) + residual
    attn_kernel<<<64 * NQTILES, 32, 0, stream>>>(qB, kB, vtB, qpoolF, attnxB);

    // 6) output projection GEMM (WMMA) -> fp32 d_out
    gemm_wmma_kernel<1><<<dim3(DIMC / 128, (M_PROJ + 127) / 128), 256, 0, stream>>>(
        (const void*)attnxB, projwB, proj_b, (void*)out, M_PROJ, DIMC);
}
